// Int4Linear_2362232013182
// MI455X (gfx1250) — matrix operand order repaired and checked by reading, not compiled, not run on hardware
//
#include <hip/hip_runtime.h>

// ---------------------------------------------------------------------------
// W8A16 GEMM for MI455X (gfx1250):  out = (x_f16 @ W_int8_f16^T) * scale + bias
//   M = B*S = 8192, N = DOUT = 16384, K = DIN = 4096   (1.10 TFLOP)
// Matrix-pipe-bound -> V_WMMA_F32_16X16X32_F16 with f32 accumulate.
// Block tile 256x128 (8 waves, wave tile 64x64 = 16 WMMAs/K-step/wave).
// Weights are exact integers |w|<8 -> exact in f16; scale folded into epilogue.
// ---------------------------------------------------------------------------

typedef __attribute__((ext_vector_type(16))) _Float16 v16h;
typedef __attribute__((ext_vector_type(8)))  _Float16 v8h;
typedef __attribute__((ext_vector_type(8)))  float    v8f;
typedef __attribute__((ext_vector_type(4)))  float    v4f;
typedef __attribute__((ext_vector_type(4)))  int      v4i;

#define DIN   4096
#define DOUT  16384
#define BM    256
#define BN    128
#define BK    32
#define LDSS  40   // padded row stride in halfs: 80B rows -> 16B aligned,
                   // 20-dword bank stride -> conflict-free across 16-lane groups

union Frag { v16h v; v8h h[2]; };

__global__ __launch_bounds__(256)
void w8a16_wmma_gemm_kernel(const float* __restrict__ x,
                            const int*   __restrict__ w,
                            const float* __restrict__ scale,
                            const float* __restrict__ bias,
                            float*       __restrict__ out)
{
    __shared__ _Float16 As[BM][LDSS];   // x tile, row-major [m][k], f16  (20 KB)
    __shared__ _Float16 Bs[BN][LDSS];   // w tile, [n][k] (K-contiguous), f16 (10 KB)

    const int tid   = threadIdx.x;
    const int lane  = tid & 31;
    const int wave  = tid >> 5;         // 0..7
    const int wm    = wave >> 1;        // 0..3  -> 64 rows of M each
    const int wn    = wave & 1;         // 0..1  -> 64 cols of N each

    const int m_blk = blockIdx.y * BM;
    const int n_blk = blockIdx.x * BN;

    // ---- global->LDS staging: chunks of 16 elements, 2 threads per row ----
    const int ld_row = tid >> 1;            // 0..127
    const int ld_off = (tid & 1) * 16;      // k offset 0 or 16

    const float* aptr0 = x + (size_t)(m_blk + ld_row) * DIN + ld_off;        // rows 0..127
    const float* aptr1 = aptr0 + (size_t)128 * DIN;                          // rows 128..255
    const int*   bptr  = w + (size_t)(n_blk + ld_row) * DIN + ld_off;

    v8f acc[4][4] = {};                     // 16x 16x16 f32 accumulators (128 VGPRs)

    const int lhalf = lane >> 4;            // 0 or 1 (lane group)
    const int lrow  = lane & 15;

    for (int kt = 0; kt < DIN; kt += BK) {
        // ---- stage A rows 0..127: fp32 -> f16 ----
        {
            v4f a0 = *(const v4f*)(aptr0 + kt);
            v4f a1 = *(const v4f*)(aptr0 + kt + 4);
            v4f a2 = *(const v4f*)(aptr0 + kt + 8);
            v4f a3 = *(const v4f*)(aptr0 + kt + 12);
            v8h h0, h1;
            #pragma unroll
            for (int i = 0; i < 4; ++i) {
                h0[i]     = (_Float16)a0[i];
                h0[4 + i] = (_Float16)a1[i];
                h1[i]     = (_Float16)a2[i];
                h1[4 + i] = (_Float16)a3[i];
            }
            *(v8h*)&As[ld_row][ld_off]     = h0;
            *(v8h*)&As[ld_row][ld_off + 8] = h1;
        }
        // ---- stage A rows 128..255 ----
        {
            v4f a0 = *(const v4f*)(aptr1 + kt);
            v4f a1 = *(const v4f*)(aptr1 + kt + 4);
            v4f a2 = *(const v4f*)(aptr1 + kt + 8);
            v4f a3 = *(const v4f*)(aptr1 + kt + 12);
            v8h h0, h1;
            #pragma unroll
            for (int i = 0; i < 4; ++i) {
                h0[i]     = (_Float16)a0[i];
                h0[4 + i] = (_Float16)a1[i];
                h1[i]     = (_Float16)a2[i];
                h1[4 + i] = (_Float16)a3[i];
            }
            *(v8h*)&As[128 + ld_row][ld_off]     = h0;
            *(v8h*)&As[128 + ld_row][ld_off + 8] = h1;
        }
        // ---- stage B: int8-valued int32 -> f16 (exact, |w| < 8) ----
        {
            v4i b0 = *(const v4i*)(bptr + kt);
            v4i b1 = *(const v4i*)(bptr + kt + 4);
            v4i b2 = *(const v4i*)(bptr + kt + 8);
            v4i b3 = *(const v4i*)(bptr + kt + 12);
            v8h h0, h1;
            #pragma unroll
            for (int i = 0; i < 4; ++i) {
                h0[i]     = (_Float16)b0[i];
                h0[4 + i] = (_Float16)b1[i];
                h1[i]     = (_Float16)b2[i];
                h1[4 + i] = (_Float16)b3[i];
            }
            *(v8h*)&Bs[ld_row][ld_off]     = h0;
            *(v8h*)&Bs[ld_row][ld_off + 8] = h1;
        }

        // prefetch next K tile into cache (global_prefetch_b8)
        if (kt + BK < DIN) {
            __builtin_prefetch((const void*)(aptr0 + kt + BK), 0, 1);
            __builtin_prefetch((const void*)(aptr1 + kt + BK), 0, 1);
            __builtin_prefetch((const void*)(bptr  + kt + BK), 0, 1);
        }

        __syncthreads();

        // ---- B fragments (reused by all 4 M-tiles) ----
        // B 32x16 f16: lane holds column N=lane%16, K = 16*(lane/16) .. +15 contiguous
        Frag bfr[4];
        #pragma unroll
        for (int nt = 0; nt < 4; ++nt) {
            const _Float16* p = &Bs[wn * 64 + nt * 16 + lrow][16 * lhalf];
            bfr[nt].h[0] = *(const v8h*)(p);
            bfr[nt].h[1] = *(const v8h*)(p + 8);
        }

        // ---- 16 WMMAs; A loaded one tile at a time to cap register pressure ----
        // A 16x32 f16: lanes 0-15 hold K=[0..15]; lanes 16-31 K=[16..31]
        #pragma unroll
        for (int mt = 0; mt < 4; ++mt) {
            Frag afr;
            const _Float16* p = &As[wm * 64 + mt * 16 + lrow][16 * lhalf];
            afr.h[0] = *(const v8h*)(p);
            afr.h[1] = *(const v8h*)(p + 8);
            #pragma unroll
            for (int nt = 0; nt < 4; ++nt)
                acc[mt][nt] = __builtin_amdgcn_wmma_f32_16x16x32_f16(
                    /*neg_a=*/false, afr.v,
                    /*neg_b=*/false, bfr[nt].v,
                    /*c_mod=*/(short)0, acc[mt][nt],
                    /*reuse_a=*/false, /*reuse_b=*/false);
        }

        __syncthreads();
    }

    // ---- epilogue: out = acc * scale[col] + bias[col] ----
    // C/D layout: VGPR r holds M = r + 8*(lane/16), N = lane%16
    #pragma unroll
    for (int nt = 0; nt < 4; ++nt) {
        const int col = n_blk + wn * 64 + nt * 16 + lrow;
        const float s = scale[col];
        const float b = bias[col];
        #pragma unroll
        for (int mt = 0; mt < 4; ++mt) {
            const int row0 = m_blk + wm * 64 + mt * 16 + 8 * lhalf;
            #pragma unroll
            for (int r = 0; r < 8; ++r) {
                out[(size_t)(row0 + r) * DOUT + col] = acc[mt][nt][r] * s + b;
            }
        }
    }
}

extern "C" void kernel_launch(void* const* d_in, const int* in_sizes, int n_in,
                              void* d_out, int out_size, void* d_ws, size_t ws_size,
                              hipStream_t stream)
{
    const float* x     = (const float*)d_in[0];   // [B,S,DIN] fp32
    const int*   w     = (const int*)  d_in[1];   // [DOUT,DIN] int32 (int8-valued)
    const float* scale = (const float*)d_in[2];   // [DOUT]
    const float* bias  = (const float*)d_in[3];   // [DOUT]
    float*       out   = (float*)d_out;           // [B,S,DOUT] fp32

    const int Mrows = in_sizes[0] / DIN;          // 8192
    dim3 grid(DOUT / BN, Mrows / BM);             // (128, 32)
    w8a16_wmma_gemm_kernel<<<grid, 256, 0, stream>>>(x, w, scale, bias, out);
}
